// FusionModel_2482491097473
// MI455X (gfx1250) — matrix operand order repaired and checked by reading, not compiled, not run on hardware
//
#include <hip/hip_runtime.h>
#include <hip/hip_bf16.h>
#include <cstdint>
#include <cstddef>

// ---------------------------------------------------------------------------
// CDNA5 (gfx1250, wave32) types
// ---------------------------------------------------------------------------
typedef __attribute__((ext_vector_type(16))) __bf16         bf16x16;
typedef __attribute__((ext_vector_type(8)))  float          f32x8;
typedef __attribute__((ext_vector_type(4)))  float          f32x4;
typedef __attribute__((ext_vector_type(8)))  unsigned short u16x8;
typedef __attribute__((ext_vector_type(4)))  unsigned short u16x4;
typedef int i32x4 __attribute__((vector_size(16)));

union Frag { u16x8 h[2]; bf16x16 v; };

// gfx1250 async global->LDS copy path (ASYNCcnt-tracked).
// Probe round 2 confirmed the builtin exists with int4* pointer params.
#if defined(__has_builtin)
#if __has_builtin(__builtin_amdgcn_global_load_async_to_lds_b128)
#define HAVE_ASYNC_COPY 1
#endif
#if __has_builtin(__builtin_amdgcn_s_wait_asynccnt)
#define HAVE_ASYNC_WAIT 1
#endif
#endif

#ifdef HAVE_ASYNC_COPY
#define ASYNC_CP_B128(g, l)                                                    \
    __builtin_amdgcn_global_load_async_to_lds_b128((i32x4*)(g), (i32x4*)(l), 0, 0)
#endif

// float -> bf16 bits (RNE)
static __device__ __forceinline__ unsigned short f2bf(float f) {
    unsigned u = __float_as_uint(f);
    u += 0x7fffu + ((u >> 16) & 1u);
    return (unsigned short)(u >> 16);
}
static __device__ __forceinline__ float bf2f(unsigned short h) {
    return __uint_as_float((unsigned)h << 16);
}

// ---------------------------------------------------------------------------
// GEMM (bf16 in, f32 accum, f32 or bf16 out):
//   C[m,n] = scale * sum_k A[m,k] * B[k,n] (+ bias[n]) (optional gelu)
//   A element (m,k) at A[m*as_m + k]          (K-contiguous)
//   B element (k,n) at B[n*bs_n + k]          (K-contiguous, i.e. N-major)
//   C element (m,n) at C[m*cs_m + n*cs_n]
// Requirements: M%64==0, N%64==0, K%32==0, as_m%8==0, bs_n%8==0 (all hold).
// 128 threads = 4 waves; 64x64 tile; K stepped by 32; double-buffered LDS;
// 4 x v_wmma_f32_16x16x32_bf16 per wave per K-step.
// ---------------------------------------------------------------------------
#define TM 64
#define TN 64
#define TK 32
#define AP 40   // LDS pitch (bf16 elems): 80B rows, 16B-aligned chunks
#define BP 40

__launch_bounds__(128)
__global__ void gemm_bf16_wmma(const unsigned short* __restrict__ A, long as_m,
                               const unsigned short* __restrict__ B, long bs_n,
                               const float* __restrict__ bias,
                               void* __restrict__ Cv, long cs_m, long cs_n,
                               int M, int N, int K, int epi, float scale,
                               int c_bf16)
{
    __shared__ __align__(16) unsigned short lA[2][TM * AP];
    __shared__ __align__(16) unsigned short lB[2][TN * BP];

    const int tid  = threadIdx.x;
    const int wave = tid >> 5;
    const int lane = tid & 31;
    const int half = lane >> 4;
    const int l15  = lane & 15;
    const int m0 = blockIdx.y * TM;
    const int n0 = blockIdx.x * TN;

    // staging geometry: each thread moves two 8-elem chunks of A and of B
    const int sr  = tid >> 2;        // 0..31 (plus +32 row)
    const int sc8 = (tid & 3) * 8;   // chunk offset within the 32-wide K tile

    auto stage = [&](int buf, int k0) {
        const unsigned short* ga0 = A + (long)(m0 + sr)      * as_m + k0 + sc8;
        const unsigned short* ga1 = A + (long)(m0 + sr + 32) * as_m + k0 + sc8;
        const unsigned short* gb0 = B + (long)(n0 + sr)      * bs_n + k0 + sc8;
        const unsigned short* gb1 = B + (long)(n0 + sr + 32) * bs_n + k0 + sc8;
        unsigned short* la0 = &lA[buf][(sr)      * AP + sc8];
        unsigned short* la1 = &lA[buf][(sr + 32) * AP + sc8];
        unsigned short* lb0 = &lB[buf][(sr)      * BP + sc8];
        unsigned short* lb1 = &lB[buf][(sr + 32) * BP + sc8];
#ifdef HAVE_ASYNC_COPY
        ASYNC_CP_B128(ga0, la0);
        ASYNC_CP_B128(ga1, la1);
        ASYNC_CP_B128(gb0, lb0);
        ASYNC_CP_B128(gb1, lb1);
#else
        if (k0 + TK < K)  // hint the next tile toward L2 (global_prefetch_b8)
            __builtin_prefetch(ga0 + (long)TK, 0, 0);
        *(u16x8*)la0 = *(const u16x8*)ga0;
        *(u16x8*)la1 = *(const u16x8*)ga1;
        *(u16x8*)lb0 = *(const u16x8*)gb0;
        *(u16x8*)lb1 = *(const u16x8*)gb1;
#endif
    };

    f32x8 acc[4] = {};
    const int arow = wave * 16 + l15;
    const int akb  = half ? 8 : 0;    // A frag K chunk base {0..7,16..23}/{8..15,24..31}
    const int bkb  = half ? 8 : 0;    // B frag K chunk base {0..7,16..23}/{8..15,24..31}

    stage(0, 0);
    int buf = 0;
    for (int k0 = 0; k0 < K; k0 += TK) {
#ifdef HAVE_ASYNC_COPY
#ifdef HAVE_ASYNC_WAIT
        __builtin_amdgcn_s_wait_asynccnt(0);
#else
        asm volatile("s_wait_asynccnt 0" ::: "memory");
#endif
#endif
        __syncthreads();
        if (k0 + TK < K) stage(buf ^ 1, k0 + TK);

        Frag af;
        af.h[0] = *(const u16x8*)&lA[buf][arow * AP + akb];
        af.h[1] = *(const u16x8*)&lA[buf][arow * AP + akb + 16];
        #pragma unroll
        for (int j = 0; j < 4; ++j) {
            Frag bfr;
            const int col = j * 16 + l15;
            bfr.h[0] = *(const u16x8*)&lB[buf][col * BP + bkb];
            bfr.h[1] = *(const u16x8*)&lB[buf][col * BP + bkb + 16];
            acc[j] = __builtin_amdgcn_wmma_f32_16x16x32_bf16(
                false, af.v, false, bfr.v, (short)0, acc[j], false, false);
        }
        buf ^= 1;
    }

    // C/D layout: lanes 0-15 -> rows 0..7 (vgpr idx = row), lanes 16-31 -> 8..15
    const int rbase = m0 + wave * 16 + (half ? 8 : 0);
    for (int j = 0; j < 4; ++j) {
        const int col = n0 + j * 16 + l15;
        const float bb = bias ? bias[col] : 0.0f;
        #pragma unroll
        for (int v = 0; v < 8; ++v) {
            float val = acc[j][v] * scale + bb;
            if (epi == 2) {  // tanh-approx gelu
                float u = 0.7978845608028654f * (val + 0.044715f * val * val * val);
                val = 0.5f * val * (1.0f + tanhf(u));
            }
            const long off = (long)(rbase + v) * cs_m + (long)col * cs_n;
            if (c_bf16) ((unsigned short*)Cv)[off] = f2bf(val);
            else        ((float*)Cv)[off] = val;
        }
    }
}

// ---------------------------------------------------------------------------
// f32 -> bf16 flat convert (n % 4 == 0)
// ---------------------------------------------------------------------------
__global__ void cvt_bf16(const float* __restrict__ x,
                         unsigned short* __restrict__ y, long n4)
{
    long i = (long)blockIdx.x * blockDim.x + threadIdx.x;
    if (i >= n4) return;
    f32x4 v = *(const f32x4*)(x + i * 4);
    u16x4 o;
    o.x = f2bf(v.x); o.y = f2bf(v.y); o.z = f2bf(v.z); o.w = f2bf(v.w);
    *(u16x4*)(y + i * 4) = o;
}

// ---------------------------------------------------------------------------
// f32 (R x C, row-major) -> bf16 transposed (C x R). R%32==0, C%32==0.
// ---------------------------------------------------------------------------
#define TT 32
__launch_bounds__(256)
__global__ void cvt_bf16_T(const float* __restrict__ x,
                           unsigned short* __restrict__ y, int R, int C)
{
    __shared__ unsigned short tile[TT][TT + 2];
    const int c0 = blockIdx.x * TT, r0 = blockIdx.y * TT;
    const int tx = threadIdx.x & 31, ty = threadIdx.x >> 5;  // 32 x 8
    #pragma unroll
    for (int i = 0; i < TT; i += 8)
        tile[ty + i][tx] = f2bf(x[(long)(r0 + ty + i) * C + (c0 + tx)]);
    __syncthreads();
    #pragma unroll
    for (int i = 0; i < TT; i += 8)
        y[(long)(c0 + ty + i) * R + (r0 + tx)] = tile[tx][ty + i];
}

// ---------------------------------------------------------------------------
// Row-wise norm: mode 0 = LayerNorm, mode 1 = RMSNorm. f32 in, bf16 out.
// y = norm(x) [*g (+b)] [*(1+emul)] [+eadd]
// ---------------------------------------------------------------------------
__launch_bounds__(256)
__global__ void rownorm(const float* __restrict__ x, long xs,
                        unsigned short* __restrict__ y, long ys,
                        const float* __restrict__ g, const float* __restrict__ b,
                        const float* __restrict__ emul, long ems,
                        const float* __restrict__ eadd, long eas,
                        int cols, int mode)
{
    __shared__ float r1[256];
    __shared__ float r2[256];
    const int row = blockIdx.x, tid = threadIdx.x;
    const float* xp = x + (long)row * xs;

    float s1 = 0.0f, s2 = 0.0f;
    for (int c = tid; c < cols; c += 256) { float v = xp[c]; s1 += v; s2 += v * v; }
    r1[tid] = s1; r2[tid] = s2;
    __syncthreads();
    for (int o = 128; o > 0; o >>= 1) {
        if (tid < o) { r1[tid] += r1[tid + o]; r2[tid] += r2[tid + o]; }
        __syncthreads();
    }
    const float invn = 1.0f / (float)cols;
    float mu, ms;
    if (mode == 0) { mu = r1[0] * invn; ms = r2[0] * invn - mu * mu; }
    else           { mu = 0.0f;         ms = r2[0] * invn; }
    const float inv = rsqrtf(ms + 1e-6f);

    unsigned short* yp = y + (long)row * ys;
    for (int c = tid; c < cols; c += 256) {
        float v = (xp[c] - mu) * inv;
        if (g)    v = v * g[c] + (b ? b[c] : 0.0f);
        if (emul) v = v * (1.0f + emul[(long)row * ems + c]);
        if (eadd) v = v + eadd[(long)row * eas + c];
        yp[c] = f2bf(v);
    }
}

// ---------------------------------------------------------------------------
// Row softmax with key-length masking: f32 scores in (scratched), bf16 out.
// ---------------------------------------------------------------------------
__launch_bounds__(256)
__global__ void softmax_rows(float* __restrict__ s,
                             unsigned short* __restrict__ p, int cols,
                             const int* __restrict__ len_ptr)
{
    __shared__ float red[256];
    const int row = blockIdx.x, tid = threadIdx.x;
    float* sp = s + (long)row * cols;
    unsigned short* pp = p + (long)row * cols;
    const int len = len_ptr ? len_ptr[0] : cols;

    float mx = -3.4e38f;
    for (int c = tid; c < cols; c += 256)
        mx = fmaxf(mx, (c < len) ? sp[c] : -1.0e9f);
    red[tid] = mx; __syncthreads();
    for (int o = 128; o > 0; o >>= 1) {
        if (tid < o) red[tid] = fmaxf(red[tid], red[tid + o]);
        __syncthreads();
    }
    mx = red[0];
    __syncthreads();

    float sum = 0.0f;
    for (int c = tid; c < cols; c += 256) {
        float v = (c < len) ? sp[c] : -1.0e9f;
        float e = __expf(v - mx);
        sp[c] = e; sum += e;
    }
    red[tid] = sum; __syncthreads();
    for (int o = 128; o > 0; o >>= 1) {
        if (tid < o) red[tid] += red[tid + o];
        __syncthreads();
    }
    const float inv = 1.0f / red[0];
    for (int c = tid; c < cols; c += 256) pp[c] = f2bf(sp[c] * inv);
}

// ---------------------------------------------------------------------------
// RoPE on bf16 (S, NH, DH)
// ---------------------------------------------------------------------------
__global__ void rope_bf16(const unsigned short* __restrict__ x,
                          unsigned short* __restrict__ y,
                          int S, int NHn, int DHn)
{
    const int half = DHn >> 1;
    const int tot  = S * NHn * half;
    int idx = blockIdx.x * blockDim.x + threadIdx.x;
    if (idx >= tot) return;
    const int d = idx % half;
    const int h = (idx / half) % NHn;
    const int s = idx / (half * NHn);
    const float inv = __expf(-(float)d / (float)half * 9.210340371976184f);
    const float ang = (float)s * inv;
    const float cs = __cosf(ang), sn = __sinf(ang);
    const long base = (long)s * NHn * DHn + (long)h * DHn;
    const float x1 = bf2f(x[base + d]);
    const float x2 = bf2f(x[base + half + d]);
    y[base + d]        = f2bf(x1 * cs - x2 * sn);
    y[base + half + d] = f2bf(x1 * sn + x2 * cs);
}

// ---------------------------------------------------------------------------
// bf16 elementwise add: out = a + b
// ---------------------------------------------------------------------------
__global__ void add_bf16(const unsigned short* __restrict__ a,
                         const unsigned short* __restrict__ b,
                         unsigned short* __restrict__ out, long n)
{
    long i = (long)blockIdx.x * blockDim.x + threadIdx.x;
    if (i >= n) return;
    out[i] = f2bf(bf2f(a[i]) + bf2f(b[i]));
}

// ---------------------------------------------------------------------------
// f32 gated residual: out = a + b * (g ? g : 1), rows x cols with row strides
// ---------------------------------------------------------------------------
__global__ void add_gate(const float* __restrict__ a, long as,
                         const float* __restrict__ b, long bs,
                         const float* __restrict__ g, long gs,
                         float* __restrict__ out, long os,
                         int rows, int cols)
{
    long idx = (long)blockIdx.x * blockDim.x + threadIdx.x;
    if (idx >= (long)rows * cols) return;
    int r = (int)(idx / cols), c = (int)(idx % cols);
    float v = a[(long)r * as + c] +
              b[(long)r * bs + c] * (g ? g[(long)r * gs + c] : 1.0f);
    out[(long)r * os + c] = v;
}

// ---------------------------------------------------------------------------
// Host orchestration
// ---------------------------------------------------------------------------
namespace {
constexpr int SVc = 2048, SAc = 1024, SCc = 512, Dc = 1024, FFNc = 4096;
constexpr int NHc = 16, DHc = 64;

enum { MOD_W = 0, MOD_B, SA_WQ, SA_BQ, SA_WK, SA_BK, SA_WV, SA_BV, SA_WO, SA_BO,
       SA_GQ, SA_GK, CA_WQ, CA_BQ, CA_WK, CA_BK, CA_WV, CA_BV, CA_WO, CA_BO,
       CA_GQ, CA_GK, KF_W, KF_B, VF_W, VF_B, KF_G, PRE_G, PRE_B, N3_G, N3_B,
       FFN_W1, FFN_B1, FFN_W2, FFN_B2 };
constexpr int PV_BASE = 10;
constexpr int PA_BASE = 45;

struct WSet {  // transposed bf16 weights (N-major, K-contiguous)
    unsigned short *mod, *saq, *sak, *sav, *sao;
    unsigned short *caq, *cak, *cav, *cao, *kf, *vf, *f1, *f2;
};
}  // namespace

extern "C" void kernel_launch(void* const* d_in, const int* in_sizes, int n_in,
                              void* d_out, int out_size, void* d_ws, size_t ws_size,
                              hipStream_t stream) {
    (void)in_sizes; (void)n_in; (void)out_size; (void)ws_size;
    typedef unsigned short ush;

    const float* vid_in = (const float*)d_in[0];
    const float* aud_in = (const float*)d_in[1];
    const float* vid_e  = (const float*)d_in[2];
    const float* aud_e  = (const float*)d_in[3];
    const float* vctx   = (const float*)d_in[4];
    const float* actx   = (const float*)d_in[5];
    const int*   vsl    = (const int*)d_in[6];
    const int*   asl    = (const int*)d_in[7];
    const int*   vcl    = (const int*)d_in[8];
    const int*   acl    = (const int*)d_in[9];

    auto P = [&](int base, int i) { return (const float*)d_in[base + i]; };

    // ---- workspace partition (256B aligned) -----------------------------
    char* wsb = (char*)d_ws;
    size_t off = 0;
    auto allocB = [&](size_t bytes) {
        void* p = wsb + off;
        off = (off + bytes + 255) & ~(size_t)255;
        return p;
    };
    auto aF = [&](size_t n) { return (float*)allocB(n * 4); };
    auto aH = [&](size_t n) { return (ush*)allocB(n * 2); };

    float* ve_mod = aF((size_t)SVc * 6 * Dc);
    float* ae_mod = aF((size_t)SAc * 6 * Dc);
    float* t1F    = aF((size_t)SVc * Dc);
    float* t2F    = aF((size_t)SVc * Dc);
    float* scoreF = aF((size_t)SVc * SVc);
    float* og_aud = aF((size_t)SAc * Dc);

    ush* veB    = aH((size_t)SVc * 6 * Dc);
    ush* aeB    = aH((size_t)SAc * 6 * Dc);
    ush* vctxB  = aH((size_t)SCc * Dc);
    ush* actxB  = aH((size_t)SCc * Dc);
    ush* xnB    = aH((size_t)SVc * Dc);
    ush* qbB    = aH((size_t)SVc * Dc);
    ush* kbB    = aH((size_t)SVc * Dc);
    ush* qrB    = aH((size_t)SVc * Dc);
    ush* krB    = aH((size_t)SVc * Dc);
    ush* vTB    = aH((size_t)SVc * Dc);   // V stored transposed (D x S)
    ush* vcTB   = aH((size_t)SCc * Dc);   // ctx V transposed (D x SC)
    ush* kcB    = aH((size_t)SCc * Dc);
    ush* attn1B = aH((size_t)SVc * Dc);
    ush* attn2B = aH((size_t)SVc * Dc);
    ush* sumB   = aH((size_t)SVc * Dc);
    ush* probB  = aH((size_t)SVc * SVc);
    ush* ffnhB  = aH((size_t)SVc * FFNc);

    WSet W[2];
    for (int s = 0; s < 2; ++s) {
        W[s].mod = aH((size_t)Dc * Dc);
        W[s].saq = aH((size_t)Dc * Dc);  W[s].sak = aH((size_t)Dc * Dc);
        W[s].sav = aH((size_t)Dc * Dc);  W[s].sao = aH((size_t)Dc * Dc);
        W[s].caq = aH((size_t)Dc * Dc);  W[s].cak = aH((size_t)Dc * Dc);
        W[s].cav = aH((size_t)Dc * Dc);  W[s].cao = aH((size_t)Dc * Dc);
        W[s].kf  = aH((size_t)Dc * Dc);  W[s].vf  = aH((size_t)Dc * Dc);
        W[s].f1  = aH((size_t)Dc * FFNc);
        W[s].f2  = aH((size_t)FFNc * Dc);
    }

    float* vid_cur = (float*)d_out;
    float* aud_cur = (float*)d_out + (size_t)SVc * Dc;

    // ---- typed launch helpers -------------------------------------------
    auto gemm = [&](const ush* A, long as_m, const ush* B, long bs_n,
                    const float* bias, void* C, long cs_m, long cs_n,
                    int M, int N, int K, int epi, float scale, int c_bf16) {
        dim3 grid((unsigned)(N / TN), (unsigned)(M / TM));
        gemm_bf16_wmma<<<grid, dim3(128), 0, stream>>>(A, as_m, B, bs_n, bias,
                                                       C, cs_m, cs_n, M, N, K,
                                                       epi, scale, c_bf16);
    };
    auto cvtT = [&](const float* x, ush* y, int R, int C) {
        cvt_bf16_T<<<dim3((unsigned)(C / TT), (unsigned)(R / TT)), 256, 0,
                     stream>>>(x, y, R, C);
    };
    auto cvtF = [&](const float* x, ush* y, long n) {
        long n4 = n / 4;
        cvt_bf16<<<dim3((unsigned)((n4 + 255) / 256)), 256, 0, stream>>>(x, y, n4);
    };
    auto norm = [&](const float* x, ush* y, const float* g, const float* b,
                    const float* emul, long ems, const float* eadd, long eas,
                    int rows, int mode) {
        rownorm<<<dim3((unsigned)rows), 256, 0, stream>>>(
            x, (long)Dc, y, (long)Dc, g, b, emul, ems, eadd, eas, Dc, mode);
    };
    auto softmax = [&](float* s, ush* p, int rows, int cols, const int* len) {
        softmax_rows<<<dim3((unsigned)rows), 256, 0, stream>>>(s, p, cols, len);
    };
    auto rope = [&](const ush* x, ush* y, int S) {
        int tot = S * NHc * (DHc / 2);
        rope_bf16<<<dim3((unsigned)((tot + 255) / 256)), 256, 0, stream>>>(
            x, y, S, NHc, DHc);
    };
    auto addh = [&](const ush* a, const ush* b, ush* o, long n) {
        add_bf16<<<dim3((unsigned)((n + 255) / 256)), 256, 0, stream>>>(a, b, o, n);
    };
    auto addg = [&](const float* a, const float* b, const float* g, long gs,
                    float* o, int S) {
        long tot = (long)S * Dc;
        add_gate<<<dim3((unsigned)((tot + 255) / 256)), 256, 0, stream>>>(
            a, (long)Dc, b, (long)Dc, g, gs, o, (long)Dc, S, Dc);
    };

    // attention: Q (Sq,NH,DH) bf16, K (Sk,NH,DH) bf16, V^T (NH*DH, Sk) bf16
    auto attention = [&](const ush* Q, const ush* Kb, const ush* VT, ush* O,
                         int Sq, int Sk, const int* len) {
        for (int h = 0; h < NHc; ++h) {
            gemm(Q + h * DHc, Dc, Kb + h * DHc, Dc, nullptr,
                 scoreF, Sk, 1, Sq, Sk, DHc, 0, 0.125f, 0);
            softmax(scoreF, probB, Sq, Sk, len);
            gemm(probB, Sk, VT + (long)h * DHc * Sk, Sk, nullptr,
                 O + h * DHc, Dc, 1, Sq, DHc, Sk, 0, 1.0f, 1);
        }
    };

    auto self_attn = [&](float* xcur, const float* e, const WSet& w, int base,
                         int S, const int* len) {
        norm(xcur, xnB, nullptr, nullptr, e + 1 * Dc, 6L * Dc, e + 0 * Dc, 6L * Dc, S, 0);
        gemm(xnB, Dc, w.saq, Dc, P(base, SA_BQ), t1F, Dc, 1, S, Dc, Dc, 0, 1.0f, 0);
        norm(t1F, qbB, P(base, SA_GQ), nullptr, nullptr, 0, nullptr, 0, S, 1);
        gemm(xnB, Dc, w.sak, Dc, P(base, SA_BK), t1F, Dc, 1, S, Dc, Dc, 0, 1.0f, 0);
        norm(t1F, kbB, P(base, SA_GK), nullptr, nullptr, 0, nullptr, 0, S, 1);
        gemm(xnB, Dc, w.sav, Dc, P(base, SA_BV), vTB, 1, S, S, Dc, Dc, 0, 1.0f, 1);
        rope(qbB, qrB, S);
        rope(kbB, krB, S);
        attention(qrB, krB, vTB, attn1B, S, S, len);
        gemm(attn1B, Dc, w.sao, Dc, P(base, SA_BO), t1F, Dc, 1, S, Dc, Dc, 0, 1.0f, 0);
        addg(xcur, t1F, e + 2 * Dc, 6L * Dc, xcur, S);
    };

    auto cross_ffn = [&](float* xcur, const ush* ctxB, const int* ctx_len,
                         const float* tgt, int St, const int* tgt_len,
                         const float* e, const WSet& w, int base, int S) {
        // q from LN(x, n3_g, n3_b)
        norm(xcur, xnB, P(base, N3_G), P(base, N3_B), nullptr, 0, nullptr, 0, S, 0);
        gemm(xnB, Dc, w.caq, Dc, P(base, CA_BQ), t1F, Dc, 1, S, Dc, Dc, 0, 1.0f, 0);
        norm(t1F, qbB, P(base, CA_GQ), nullptr, nullptr, 0, nullptr, 0, S, 1);
        // context k/v
        gemm(ctxB, Dc, w.cak, Dc, P(base, CA_BK), t2F, Dc, 1, SCc, Dc, Dc, 0, 1.0f, 0);
        norm(t2F, kcB, P(base, CA_GK), nullptr, nullptr, 0, nullptr, 0, SCc, 1);
        gemm(ctxB, Dc, w.cav, Dc, P(base, CA_BV), vcTB, 1, SCc, SCc, Dc, Dc, 0, 1.0f, 1);
        attention(qbB, kcB, vcTB, attn1B, S, SCc, ctx_len);
        // target fusion path
        norm(tgt, xnB, P(base, PRE_G), P(base, PRE_B), nullptr, 0, nullptr, 0, St, 0);
        gemm(xnB, Dc, w.kf, Dc, P(base, KF_B), t1F, Dc, 1, St, Dc, Dc, 0, 1.0f, 0);
        norm(t1F, kbB, P(base, KF_G), nullptr, nullptr, 0, nullptr, 0, St, 1);
        gemm(xnB, Dc, w.vf, Dc, P(base, VF_B), vTB, 1, St, St, Dc, Dc, 0, 1.0f, 1);
        rope(qbB, qrB, S);
        rope(kbB, krB, St);
        attention(qrB, krB, vTB, attn2B, S, St, tgt_len);
        // out projection of (x1 + x2)
        addh(attn1B, attn2B, sumB, (long)S * Dc);
        gemm(sumB, Dc, w.cao, Dc, P(base, CA_BO), t2F, Dc, 1, S, Dc, Dc, 0, 1.0f, 0);
        addg(xcur, t2F, nullptr, 0, xcur, S);
        // FFN with AdaLN modulation
        norm(xcur, xnB, nullptr, nullptr, e + 4 * Dc, 6L * Dc, e + 3 * Dc, 6L * Dc, S, 0);
        gemm(xnB, Dc, w.f1, Dc, P(base, FFN_B1), ffnhB, FFNc, 1, S, FFNc, Dc, 2, 1.0f, 1);
        gemm(ffnhB, FFNc, w.f2, FFNc, P(base, FFN_B2), t1F, Dc, 1, S, Dc, FFNc, 0, 1.0f, 0);
        addg(xcur, t1F, e + 5 * Dc, 6L * Dc, xcur, S);
    };

    // ---- one-time conversions (weights transposed to N-major bf16) ------
    const int bases[2] = {PV_BASE, PA_BASE};
    for (int s = 0; s < 2; ++s) {
        int b = bases[s];
        cvtT(P(b, MOD_W), W[s].mod, Dc, Dc);
        cvtT(P(b, SA_WQ), W[s].saq, Dc, Dc);
        cvtT(P(b, SA_WK), W[s].sak, Dc, Dc);
        cvtT(P(b, SA_WV), W[s].sav, Dc, Dc);
        cvtT(P(b, SA_WO), W[s].sao, Dc, Dc);
        cvtT(P(b, CA_WQ), W[s].caq, Dc, Dc);
        cvtT(P(b, CA_WK), W[s].cak, Dc, Dc);
        cvtT(P(b, CA_WV), W[s].cav, Dc, Dc);
        cvtT(P(b, CA_WO), W[s].cao, Dc, Dc);
        cvtT(P(b, KF_W),  W[s].kf,  Dc, Dc);
        cvtT(P(b, VF_W),  W[s].vf,  Dc, Dc);
        cvtT(P(b, FFN_W1), W[s].f1, Dc, FFNc);
        cvtT(P(b, FFN_W2), W[s].f2, FFNc, Dc);
    }
    cvtF(vid_e, veB, (long)SVc * 6 * Dc);
    cvtF(aud_e, aeB, (long)SAc * 6 * Dc);
    cvtF(vctx, vctxB, (long)SCc * Dc);
    cvtF(actx, actxB, (long)SCc * Dc);

    (void)hipMemcpyAsync(vid_cur, vid_in, (size_t)SVc * Dc * 4,
                         hipMemcpyDeviceToDevice, stream);
    (void)hipMemcpyAsync(aud_cur, aud_in, (size_t)SAc * Dc * 4,
                         hipMemcpyDeviceToDevice, stream);

    // modulation: e @ mod_w + mod_b
    gemm(aeB, Dc, W[1].mod, Dc, P(PA_BASE, MOD_B), ae_mod, Dc, 1,
         SAc * 6, Dc, Dc, 0, 1.0f, 0);
    gemm(veB, Dc, W[0].mod, Dc, P(PV_BASE, MOD_B), ve_mod, Dc, 1,
         SVc * 6, Dc, Dc, 0, 1.0f, 0);

    self_attn(aud_cur, ae_mod, W[1], PA_BASE, SAc, asl);
    self_attn(vid_cur, ve_mod, W[0], PV_BASE, SVc, vsl);

    (void)hipMemcpyAsync(og_aud, aud_cur, (size_t)SAc * Dc * 4,
                         hipMemcpyDeviceToDevice, stream);

    cross_ffn(aud_cur, actxB, acl, vid_cur, SVc, vsl, ae_mod, W[1], PA_BASE, SAc);
    cross_ffn(vid_cur, vctxB, vcl, og_aud, SAc, asl, ve_mod, W[0], PV_BASE, SVc);
}
